// LinearAttentionLayer_48498770706751
// MI455X (gfx1250) — compile-verified, matrix operand order checked by reading, not run on hardware
//
#include <hip/hip_runtime.h>
#include <hip/hip_bf16.h>

// ---------------------------------------------------------------------------
// LinearAttention (B=4,S=4096,H=1024,NH=16,HD=64) for gfx1250 (MI455X).
// Heavy math via v_wmma_f32_16x16x32_bf16; tile staging via
// GLOBAL_LOAD_ASYNC_TO_LDS_B128 (ASYNCcnt) with double buffering.
// ---------------------------------------------------------------------------

#define BB 4
#define SS 4096
#define HH 1024
#define NHEAD 16
#define HDIM 64

typedef __attribute__((ext_vector_type(16))) __bf16 v16bf;
typedef __attribute__((ext_vector_type(8)))  __bf16 v8bf;
typedef __attribute__((ext_vector_type(8)))  float  v8f;
typedef __attribute__((ext_vector_type(4)))  unsigned int v4u;
typedef __attribute__((ext_vector_type(4)))  int v4i;

union FragBF {
    v16bf v;
    v8bf  h[2];
};

#if __has_builtin(__builtin_amdgcn_global_load_async_to_lds_b128) && \
    __has_builtin(__builtin_amdgcn_s_wait_asynccnt)
#define HAVE_ASYNC_LDS 1
#endif

using glb_v4i = __attribute__((address_space(1))) v4i*;
using lds_v4i = __attribute__((address_space(3))) v4i*;

// 16-byte global -> LDS copy. Async (no VGPR staging) when available.
__device__ __forceinline__ void cp16_g2l(const unsigned short* g, unsigned short* l) {
#ifdef HAVE_ASYNC_LDS
    __builtin_amdgcn_global_load_async_to_lds_b128(
        reinterpret_cast<glb_v4i>((unsigned long long)g),
        reinterpret_cast<lds_v4i>((unsigned long long)l),
        0, 0);
#else
    *reinterpret_cast<v8bf*>(l) = *reinterpret_cast<const v8bf*>(g);
#endif
}

#ifdef HAVE_ASYNC_LDS
#define WAIT_ASYNC(n) __builtin_amdgcn_s_wait_asynccnt(n)
#else
#define WAIT_ASYNC(n)
#endif

__device__ __forceinline__ unsigned short f2bf(float f) {
    unsigned int u = __float_as_uint(f);
    u += 0x7fffu + ((u >> 16) & 1u);          // round-to-nearest-even
    return (unsigned short)(u >> 16);
}

__device__ __forceinline__ float act_elu(float x) {
    return x > 0.0f ? x : (__expf(x) - 1.0f);
}

__device__ __forceinline__ v8f wmma_bf16(v16bf a, v16bf b, v8f c) {
    // (neg_a, A, neg_b, B, c_mod, C, reuse_a, reuse_b)
    return __builtin_amdgcn_wmma_f32_16x16x32_bf16(false, a, false, b,
                                                   (short)0, c, false, false);
}

// ---------------------------------------------------------------------------
// fp32 -> bf16 conversion (vectorized, n divisible by 1024)
// ---------------------------------------------------------------------------
__global__ __launch_bounds__(256)
void convert_f32_bf16(const float* __restrict__ in, unsigned short* __restrict__ out, int n) {
    int i = (blockIdx.x * 256 + threadIdx.x) * 4;
    if (i + 3 < n) {
        float4 f = *reinterpret_cast<const float4*>(&in[i]);
        unsigned int p0 = (unsigned int)f2bf(f.x) | ((unsigned int)f2bf(f.y) << 16);
        unsigned int p1 = (unsigned int)f2bf(f.z) | ((unsigned int)f2bf(f.w) << 16);
        uint2 p = {p0, p1};
        *reinterpret_cast<uint2*>(&out[i]) = p;
    }
}

// ---------------------------------------------------------------------------
// Generic NT GEMM:  C[m,n] = act( sum_k A[m,k] * Bt[n,k] )
// 256 threads = 8 waves; block tile 128(M) x 64(N); wave w owns rows
// [w*16, w*16+16) x all 64 N columns (4 accumulators).
// Double-buffered LDS filled with async global->LDS b128 copies:
// 3 async loads / thread / stage  =>  s_wait_asynccnt 3 overlaps stages.
// ACT: 0 none, 1 elu.
// OUT: 0 row-major bf16, 1 transposed bf16 Ct[(b*Ntot+n)*Sld + s], 2 row-major f32
// ---------------------------------------------------------------------------
template<int ACT, int OUT>
__global__ __launch_bounds__(256)
void gemm_nt(const unsigned short* __restrict__ A,
             const unsigned short* __restrict__ Bt,
             void* __restrict__ Cout,
             int Kd, int lda, int ldb, int ldc, int Ntot, int Sld)
{
    __shared__ unsigned short As[2][128][40];   // +8 pad: conflict-free b128 reads
    __shared__ unsigned short Bs[2][64][40];

    const int tid  = threadIdx.x;
    const int lane = tid & 31;
    const int w    = tid >> 5;
    const int lr   = lane & 15;
    const int lh   = lane >> 4;
    const int m_blk = blockIdx.x * 128;
    const int n_blk = blockIdx.y * 64;

    const int crow = tid >> 2;          // 0..63
    const int ccol = (tid & 3) * 8;     // 0/8/16/24 (elements)

    v8f acc[4] = {};

    auto stage = [&](int buf, int k0) {
        // A tile: 128 rows x 32 cols = 512 x 16B chunks; this thread: rows crow, crow+64
        cp16_g2l(&A[(size_t)(m_blk + crow) * lda + k0 + ccol],      &As[buf][crow][ccol]);
        cp16_g2l(&A[(size_t)(m_blk + crow + 64) * lda + k0 + ccol], &As[buf][crow + 64][ccol]);
        // B tile: 64 rows x 32 cols = 256 x 16B chunks; one per thread
        cp16_g2l(&Bt[(size_t)(n_blk + crow) * ldb + k0 + ccol],     &Bs[buf][crow][ccol]);
    };

    const int nsteps = Kd >> 5;
    stage(0, 0);
    for (int kt = 0; kt < nsteps; ++kt) {
        const int cur = kt & 1;
        if (kt + 1 < nsteps) {
            stage(cur ^ 1, (kt + 1) * 32);
            WAIT_ASYNC(3);      // current buffer's 3 copies done; next 3 in flight
        } else {
            WAIT_ASYNC(0);
        }
        __syncthreads();

        FragBF a;
        a.h[0] = *reinterpret_cast<const v8bf*>(&As[cur][w * 16 + lr][lh * 16]);
        a.h[1] = *reinterpret_cast<const v8bf*>(&As[cur][w * 16 + lr][lh * 16 + 8]);
#pragma unroll
        for (int j = 0; j < 4; ++j) {
            FragBF bfr;
            bfr.h[0] = *reinterpret_cast<const v8bf*>(&Bs[cur][j * 16 + lr][lh * 16]);
            bfr.h[1] = *reinterpret_cast<const v8bf*>(&Bs[cur][j * 16 + lr][lh * 16 + 8]);
            acc[j] = wmma_bf16(a.v, bfr.v, acc[j]);
        }
        __syncthreads();        // consumers done before this buffer is refilled
    }

    // ---------------- epilogue ----------------
    const int m_base = m_blk + w * 16 + lh * 8;

    if (OUT == 0) {  // row-major bf16
        unsigned short* C = reinterpret_cast<unsigned short*>(Cout);
#pragma unroll
        for (int j = 0; j < 4; ++j) {
            const int n = n_blk + j * 16 + lr;
#pragma unroll
            for (int r = 0; r < 8; ++r) {
                float x = acc[j][r];
                if (ACT) x = act_elu(x);
                C[(size_t)(m_base + r) * ldc + n] = f2bf(x);
            }
        }
    } else if (OUT == 2) {  // row-major fp32 (final output)
        float* C = reinterpret_cast<float*>(Cout);
#pragma unroll
        for (int j = 0; j < 4; ++j) {
            const int n = n_blk + j * 16 + lr;
#pragma unroll
            for (int r = 0; r < 8; ++r) {
                float x = acc[j][r];
                if (ACT) x = act_elu(x);
                C[(size_t)(m_base + r) * ldc + n] = x;
            }
        }
    } else {  // OUT == 1: transposed bf16, Ct[(b*Ntot + n)*Sld + s]; one b128 store/lane/tile
        unsigned short* C = reinterpret_cast<unsigned short*>(Cout);
        const int b = m_blk / Sld;
        const int s_base = (m_blk % Sld) + w * 16 + lh * 8;
#pragma unroll
        for (int j = 0; j < 4; ++j) {
            const int n = n_blk + j * 16 + lr;
            unsigned short e[8];
#pragma unroll
            for (int r = 0; r < 8; ++r) {
                float x = acc[j][r];
                if (ACT) x = act_elu(x);
                e[r] = f2bf(x);
            }
            v4u p;
            p.x = (unsigned int)e[0] | ((unsigned int)e[1] << 16);
            p.y = (unsigned int)e[2] | ((unsigned int)e[3] << 16);
            p.z = (unsigned int)e[4] | ((unsigned int)e[5] << 16);
            p.w = (unsigned int)e[6] | ((unsigned int)e[7] << 16);
            *reinterpret_cast<v4u*>(&C[((size_t)b * Ntot + n) * Sld + s_base]) = p;
        }
    }
}

// ---------------------------------------------------------------------------
// kv state: kvT[bh][e][d] = sum_s Vt[bh*64+e][s] * Kt[bh*64+d][s]
// One block per (b,h): async double-buffered LDS staging of 64x32 V and K
// tiles (2 async b128 / thread / stage); 8 waves, each a 16(e) x 32(d) tile.
// ---------------------------------------------------------------------------
__global__ __launch_bounds__(256)
void kv_state(const unsigned short* __restrict__ Vt,
              const unsigned short* __restrict__ Kt,
              unsigned short* __restrict__ kvT)
{
    __shared__ unsigned short Vs[2][64][40];
    __shared__ unsigned short Ks[2][64][40];

    const int bh   = blockIdx.x;        // b*16+h
    const int bat  = bh >> 4;
    const int head = bh & 15;
    const int tid  = threadIdx.x;
    const int lane = tid & 31;
    const int w    = tid >> 5;
    const int lr   = lane & 15;
    const int lh   = lane >> 4;

    const int e_tile  = (w >> 1) * 16;  // 0/16/32/48
    const int d_tile0 = (w & 1) * 32;   // 0/32

    const int crow = tid >> 2;          // 0..63
    const int ccol = (tid & 3) * 8;     // elements

    const size_t base = ((size_t)bat * HH + head * HDIM + crow) * SS;

    auto stage = [&](int buf, int s0) {
        cp16_g2l(&Vt[base + s0 + ccol], &Vs[buf][crow][ccol]);
        cp16_g2l(&Kt[base + s0 + ccol], &Ks[buf][crow][ccol]);
    };

    v8f acc0 = {}, acc1 = {};
    stage(0, 0);
    const int nsteps = SS >> 5;  // 128
    for (int kt = 0; kt < nsteps; ++kt) {
        const int cur = kt & 1;
        if (kt + 1 < nsteps) {
            stage(cur ^ 1, (kt + 1) * 32);
            WAIT_ASYNC(2);
        } else {
            WAIT_ASYNC(0);
        }
        __syncthreads();

        FragBF a, b0, b1;
        a.h[0]  = *reinterpret_cast<const v8bf*>(&Vs[cur][e_tile + lr][lh * 16]);
        a.h[1]  = *reinterpret_cast<const v8bf*>(&Vs[cur][e_tile + lr][lh * 16 + 8]);
        b0.h[0] = *reinterpret_cast<const v8bf*>(&Ks[cur][d_tile0 + lr][lh * 16]);
        b0.h[1] = *reinterpret_cast<const v8bf*>(&Ks[cur][d_tile0 + lr][lh * 16 + 8]);
        b1.h[0] = *reinterpret_cast<const v8bf*>(&Ks[cur][d_tile0 + 16 + lr][lh * 16]);
        b1.h[1] = *reinterpret_cast<const v8bf*>(&Ks[cur][d_tile0 + 16 + lr][lh * 16 + 8]);
        acc0 = wmma_bf16(a.v, b0.v, acc0);
        acc1 = wmma_bf16(a.v, b1.v, acc1);
        __syncthreads();
    }

    unsigned short* out = kvT + (size_t)bh * HDIM * HDIM;
#pragma unroll
    for (int r = 0; r < 8; ++r) {
        const int e = e_tile + lh * 8 + r;
        out[e * HDIM + d_tile0 + lr]      = f2bf(acc0[r]);
        out[e * HDIM + d_tile0 + 16 + lr] = f2bf(acc1[r]);
    }
}

// ---------------------------------------------------------------------------
// ctx[s, h*64+e] = sum_d Q[s, h*64+d] * kvT[bh][e][d]   (per b,h; K=64)
// grid (S/128, B*NH); 8 waves, wave w: rows [m0+w*16, ..), all 64 e columns.
// kvT is L2-resident (0.5 MB); Q is streamed -> direct global loads.
// ---------------------------------------------------------------------------
__global__ __launch_bounds__(256)
void ctx_apply(const unsigned short* __restrict__ Q,
               const unsigned short* __restrict__ kvT,
               unsigned short* __restrict__ ctx)
{
    const int bh   = blockIdx.y;
    const int bat  = bh >> 4;
    const int head = bh & 15;
    const int m0   = blockIdx.x * 128;
    const int tid  = threadIdx.x;
    const int lane = tid & 31;
    const int w    = tid >> 5;
    const int lr   = lane & 15;
    const int lh   = lane >> 4;

    const unsigned short* Abase = Q + ((size_t)bat * SS) * HH + head * HDIM;
    const unsigned short* Bt    = kvT + (size_t)bh * HDIM * HDIM;  // [e][d]

    v8f acc[4] = {};
    const int arow = m0 + w * 16 + lr;
#pragma unroll
    for (int kk = 0; kk < 2; ++kk) {
        const int k0 = kk * 32 + lh * 16;
        FragBF a;
        a.h[0] = *reinterpret_cast<const v8bf*>(&Abase[(size_t)arow * HH + k0]);
        a.h[1] = *reinterpret_cast<const v8bf*>(&Abase[(size_t)arow * HH + k0 + 8]);
#pragma unroll
        for (int j = 0; j < 4; ++j) {
            FragBF bfr;
            bfr.h[0] = *reinterpret_cast<const v8bf*>(&Bt[(j * 16 + lr) * HDIM + k0]);
            bfr.h[1] = *reinterpret_cast<const v8bf*>(&Bt[(j * 16 + lr) * HDIM + k0 + 8]);
            acc[j] = wmma_bf16(a.v, bfr.v, acc[j]);
        }
    }

    unsigned short* C = ctx + ((size_t)bat * SS) * HH + head * HDIM;
    const int m_base = m0 + w * 16 + lh * 8;
#pragma unroll
    for (int j = 0; j < 4; ++j) {
#pragma unroll
        for (int r = 0; r < 8; ++r)
            C[(size_t)(m_base + r) * HH + j * 16 + lr] = f2bf(acc[j][r]);
    }
}

// ---------------------------------------------------------------------------
extern "C" void kernel_launch(void* const* d_in, const int* in_sizes, int n_in,
                              void* d_out, int out_size, void* d_ws, size_t ws_size,
                              hipStream_t stream) {
    (void)in_sizes; (void)n_in; (void)out_size; (void)ws_size;

    const float* x  = (const float*)d_in[0];
    const float* Wq = (const float*)d_in[1];
    const float* Wk = (const float*)d_in[2];
    const float* Wv = (const float*)d_in[3];
    const float* Wo = (const float*)d_in[4];

    char* ws = (char*)d_ws;
    const size_t MB = 1ull << 20;
    unsigned short* Xb   = (unsigned short*)(ws);              // 32 MB  [B*S, H]
    unsigned short* Wqb  = (unsigned short*)(ws + 32 * MB);    // 2 MB
    unsigned short* Wkb  = (unsigned short*)(ws + 34 * MB);    // 2 MB
    unsigned short* Wvb  = (unsigned short*)(ws + 36 * MB);    // 2 MB
    unsigned short* Wob  = (unsigned short*)(ws + 38 * MB);    // 2 MB
    unsigned short* Qb   = (unsigned short*)(ws + 40 * MB);    // 32 MB [B*S, H]
    unsigned short* Ktb  = (unsigned short*)(ws + 72 * MB);    // 32 MB [B, H, S]
    unsigned short* Vtb  = (unsigned short*)(ws + 104 * MB);   // 32 MB [B, H, S]
    unsigned short* kvTb = (unsigned short*)(ws + 136 * MB);   // 0.5 MB [B*NH, 64, 64]
    unsigned short* Ctxb = (unsigned short*)(ws + 137 * MB);   // 32 MB [B*S, H]

    const int nX = BB * SS * HH;   // 16,777,216
    const int nW = HH * HH;        // 1,048,576

    convert_f32_bf16<<<nX / 1024, 256, 0, stream>>>(x,  Xb,  nX);
    convert_f32_bf16<<<nW / 1024, 256, 0, stream>>>(Wq, Wqb, nW);
    convert_f32_bf16<<<nW / 1024, 256, 0, stream>>>(Wk, Wkb, nW);
    convert_f32_bf16<<<nW / 1024, 256, 0, stream>>>(Wv, Wvb, nW);
    convert_f32_bf16<<<nW / 1024, 256, 0, stream>>>(Wo, Wob, nW);

    const dim3 gproj((BB * SS) / 128, HH / 64);  // (128, 16)

    // Q = elu(X Wq^T), row-major bf16
    gemm_nt<1, 0><<<gproj, 256, 0, stream>>>(Xb, Wqb, Qb, HH, HH, HH, HH, HH, SS);
    // Kt = elu(X Wk^T) stored transposed [B,H,S]
    gemm_nt<1, 1><<<gproj, 256, 0, stream>>>(Xb, Wkb, Ktb, HH, HH, HH, HH, HH, SS);
    // Vt = (X Wv^T) stored transposed [B,H,S]
    gemm_nt<0, 1><<<gproj, 256, 0, stream>>>(Xb, Wvb, Vtb, HH, HH, HH, HH, HH, SS);

    // kvT[bh][e][d] = sum_s Vt[e,s] Kt[d,s]
    kv_state<<<BB * NHEAD, 256, 0, stream>>>(Vtb, Ktb, kvTb);

    // ctx = Q @ kvT^T per head
    ctx_apply<<<dim3(SS / 128, BB * NHEAD), 256, 0, stream>>>(Qb, kvTb, Ctxb);

    // y = ctx @ Wo^T, fp32 output
    gemm_nt<0, 2><<<gproj, 256, 0, stream>>>(Ctxb, Wob, d_out, HH, HH, HH, HH, HH, SS);
}
